// SpatialAttentionLSTMDecoder_22565758173557
// MI455X (gfx1250) — compile-verified, matrix operand order checked by reading, not run on hardware
//
#include <hip/hip_runtime.h>
#include <hip/hip_bf16.h>

typedef __bf16 bf16;
typedef __attribute__((ext_vector_type(8)))  __bf16 bf16x8;
typedef __attribute__((ext_vector_type(16))) __bf16 bf16x16;
typedef __attribute__((ext_vector_type(8)))  float  f32x8;

// gfx1250 async global->LDS path (ASYNCcnt), with compile-safe fallback.
// Probe result: param0 of the builtin is 'int __attribute__((vector_size(16))) __device__ *'
// i.e. a non-const AS1 pointer to int4; param1 is the LDS-side (AS3) pointer.
#if defined(__AMDGCN__) && __has_builtin(__builtin_amdgcn_global_load_async_to_lds_b128) && __has_builtin(__builtin_amdgcn_s_wait_asynccnt)
#define USE_ASYNC_LDS 1
typedef __attribute__((ext_vector_type(4))) int int4v;
typedef __attribute__((address_space(1))) int4v* as1_v4p;
typedef __attribute__((address_space(3))) int4v* as3_v4p;
__device__ __forceinline__ void async_cp16(const void* g, void* l) {
    __builtin_amdgcn_global_load_async_to_lds_b128((as1_v4p)g, (as3_v4p)l, 0, 0);
}
#else
#define USE_ASYNC_LDS 0
#endif

#define BM 64
#define BN 128
#define BK 64
#define LDSP (BK + 8)   // LDS row stride (bf16): +16B pad, keeps b128 alignment

// ---------------------------------------------------------------------------
// bf16 WMMA GEMM:  C[M,N] (f32) = A[M,K] * B[N,K]^T (+ bias[N])
// Requirements (all call sites satisfy them):
//   M % 64 == 0, K % 64 == 0, rows of A/B 16B aligned.
//   B may be over-read up to 127 rows past N (caller pads allocation).
// Block tile 64x128, K-step 64, 8 waves; each wave owns a 32x32 patch ->
// 8 x v_wmma_f32_16x16x32_bf16 per staged tile, double-buffered LDS with
// async global->LDS fill overlapped with compute when available.
// ---------------------------------------------------------------------------
__global__ __launch_bounds__(256) void gemm_bf16_wmma(
    const bf16* __restrict__ A, int lda,
    const bf16* __restrict__ B, int ldb,
    float* __restrict__ C, int ldc,
    const float* __restrict__ bias,
    int M, int N, int K)
{
    __shared__ bf16 As[2][BM][LDSP];
    __shared__ bf16 Bs[2][BN][LDSP];

    const int bm   = blockIdx.y * BM;
    const int bn   = blockIdx.x * BN;
    const int tid  = threadIdx.x;
    const int wave = tid >> 5;
    const int lane = tid & 31;
    const int tr   = (wave & 1) * 32;   // wave's M offset in block tile
    const int tc   = (wave >> 1) * 32;  // wave's N offset in block tile

    // staging assignment: A tile 64x64 -> thread = (row, 16-elem chunk),
    //                     B tile 128x64 -> thread = (row, 32-elem chunk)
    const int a_r = tid >> 2;
    const int a_c = (tid & 3) * 16;
    const int b_r = tid >> 1;
    const int b_c = (tid & 1) * 32;

    const bf16* gA = A + (long)(bm + a_r) * lda + a_c;
    const bf16* gB = B + (long)(bn + b_r) * ldb + b_c;

    f32x8 acc00 = {}, acc01 = {}, acc10 = {}, acc11 = {};
    const int nk = K / BK;

    auto stage = [&](int buf) {
#if USE_ASYNC_LDS
        async_cp16(gA,      &As[buf][a_r][a_c]);
        async_cp16(gA + 8,  &As[buf][a_r][a_c + 8]);
        async_cp16(gB,      &Bs[buf][b_r][b_c]);
        async_cp16(gB + 8,  &Bs[buf][b_r][b_c + 8]);
        async_cp16(gB + 16, &Bs[buf][b_r][b_c + 16]);
        async_cp16(gB + 24, &Bs[buf][b_r][b_c + 24]);
#else
        bf16x8 a0 = *(const bf16x8*)(gA);
        bf16x8 a1 = *(const bf16x8*)(gA + 8);
        bf16x8 b0 = *(const bf16x8*)(gB);
        bf16x8 b1 = *(const bf16x8*)(gB + 8);
        bf16x8 b2 = *(const bf16x8*)(gB + 16);
        bf16x8 b3 = *(const bf16x8*)(gB + 24);
        __builtin_prefetch(gA + 2 * BK, 0, 1);
        __builtin_prefetch(gB + 2 * BK, 0, 1);
        *(bf16x8*)&As[buf][a_r][a_c]      = a0;
        *(bf16x8*)&As[buf][a_r][a_c + 8]  = a1;
        *(bf16x8*)&Bs[buf][b_r][b_c]      = b0;
        *(bf16x8*)&Bs[buf][b_r][b_c + 8]  = b1;
        *(bf16x8*)&Bs[buf][b_r][b_c + 16] = b2;
        *(bf16x8*)&Bs[buf][b_r][b_c + 24] = b3;
#endif
        gA += BK;
        gB += BK;
    };

    // fragment layout (ISA 16-bit A/B): lanes 0-15 K-octets {0-7,16-23},
    // lanes 16-31 K-octets {8-15,24-31}
    const int rsel = lane & 15;
    const int kb   = (lane >> 4) * 8;

    auto loadfrag = [&](const bf16* p) -> bf16x16 {
        bf16x8 lo = *(const bf16x8*)(p + kb);
        bf16x8 hi = *(const bf16x8*)(p + kb + 16);
        bf16x16 r;
        #pragma unroll
        for (int i = 0; i < 8; ++i) { r[i] = lo[i]; r[8 + i] = hi[i]; }
        return r;
    };

    stage(0);
#if USE_ASYNC_LDS
    __builtin_amdgcn_s_wait_asynccnt(0);
#endif
    __syncthreads();

    for (int kt = 0; kt < nk; ++kt) {
        const int cur = kt & 1;
        if (kt + 1 < nk) stage(cur ^ 1);   // fill other buffer while computing

        #pragma unroll
        for (int ks = 0; ks < BK; ks += 32) {
            bf16x16 a0 = loadfrag(&As[cur][tr + rsel][ks]);
            bf16x16 a1 = loadfrag(&As[cur][tr + 16 + rsel][ks]);
            bf16x16 b0 = loadfrag(&Bs[cur][tc + rsel][ks]);
            bf16x16 b1 = loadfrag(&Bs[cur][tc + 16 + rsel][ks]);
            acc00 = __builtin_amdgcn_wmma_f32_16x16x32_bf16(false, a0, false, b0,
                                                            (short)0, acc00, false, false);
            acc01 = __builtin_amdgcn_wmma_f32_16x16x32_bf16(false, a0, false, b1,
                                                            (short)0, acc01, false, false);
            acc10 = __builtin_amdgcn_wmma_f32_16x16x32_bf16(false, a1, false, b0,
                                                            (short)0, acc10, false, false);
            acc11 = __builtin_amdgcn_wmma_f32_16x16x32_bf16(false, a1, false, b1,
                                                            (short)0, acc11, false, false);
        }
#if USE_ASYNC_LDS
        __builtin_amdgcn_s_wait_asynccnt(0);
#endif
        __syncthreads();
    }

    // C/D layout: lane L -> col (L&15); VGPR r -> row (L>>4)*8 + r
    const int rbase = (lane >> 4) * 8;
    #pragma unroll
    for (int i = 0; i < 2; ++i) {
        #pragma unroll
        for (int j = 0; j < 2; ++j) {
            f32x8 acc = i ? (j ? acc11 : acc10) : (j ? acc01 : acc00);
            const int col = bn + tc + j * 16 + rsel;
            if (col < N) {
                const float bv = bias ? bias[col] : 0.0f;
                #pragma unroll
                for (int r = 0; r < 8; ++r) {
                    const int row = bm + tr + i * 16 + rbase + r;  // M%64==0 -> in range
                    C[(long)row * ldc + col] = acc[r] + bv;
                }
            }
        }
    }
}

// ---------------------------------------------------------------------------
// Small helper kernels
// ---------------------------------------------------------------------------
__global__ void k_cvt_rows(const float* __restrict__ src, bf16* __restrict__ dst,
                           long total, int cols, int dstStride, int dstOff)
{
    long i = (long)blockIdx.x * blockDim.x + threadIdx.x;
    if (i >= total) return;
    long r = i / cols;
    int  c = (int)(i - r * cols);
    dst[r * (long)dstStride + dstOff + c] = (bf16)src[i];
}

__global__ void k_bias_add(const float* __restrict__ a, const float* __restrict__ b,
                           float* __restrict__ out, int n)
{
    int i = blockIdx.x * blockDim.x + threadIdx.x;
    if (i < n) out[i] = a[i] + b[i];
}

__global__ void k_zero_state(float* __restrict__ h, float* __restrict__ c,
                             bf16* __restrict__ hbf, int n)
{
    int i = blockIdx.x * blockDim.x + threadIdx.x;
    if (i < n) { h[i] = 0.0f; c[i] = 0.0f; hbf[i] = (bf16)0.0f; }
}

__global__ void k_embed_gather(const float* __restrict__ embed_w,
                               const int* __restrict__ cap,
                               bf16* __restrict__ embT)   // [T,B,H] bf16
{
    long i = (long)blockIdx.x * blockDim.x + threadIdx.x;
    if (i >= (long)32 * 128 * 512) return;
    int  hcol = (int)(i & 511);
    long tb   = i >> 9;
    int  b    = (int)(tb & 127);
    int  t    = (int)(tb >> 7);
    int  tok  = cap[b * 32 + t];
    embT[i] = (bf16)embed_w[(long)tok * 512 + hcol];
}

// Bahdanau attention, one batch row per block:
// scores_n = dot(tanh(proj[b,n,:] + ph[b,:]), w_score); softmax over n=49;
// ctx written as bf16 directly into x_cat cols [512, 2560).
__global__ __launch_bounds__(256) void k_attn(
    const float* __restrict__ proj,      // [B*49, 512]
    const float* __restrict__ ph,        // [B, 512]
    const float* __restrict__ wscore,    // [512]
    const float* __restrict__ features,  // [B, 49, 2048]
    bf16* __restrict__ xcat)             // [B, 3072]
{
    const int b    = blockIdx.x;
    const int tid  = threadIdx.x;
    const int wave = tid >> 5, lane = tid & 31;

    __shared__ float s_ph[512];
    __shared__ float s_sc[64];
    __shared__ float s_alpha[64];

    for (int i = tid; i < 512; i += 256) s_ph[i] = ph[b * 512 + i];
    __syncthreads();

    for (int n = wave; n < 49; n += 8) {
        const float* pf = proj + ((long)b * 49 + n) * 512;
        float acc = 0.0f;
        for (int h = lane; h < 512; h += 32)
            acc += tanhf(pf[h] + s_ph[h]) * wscore[h];
        #pragma unroll
        for (int off = 16; off > 0; off >>= 1) acc += __shfl_down(acc, off);
        if (lane == 0) s_sc[n] = acc;
    }
    __syncthreads();

    if (wave == 0) {
        float v0 = (lane < 49)      ? s_sc[lane]      : -3.0e38f;
        float v1 = (lane + 32 < 49) ? s_sc[lane + 32] : -3.0e38f;
        float m = fmaxf(v0, v1);
        #pragma unroll
        for (int off = 16; off > 0; off >>= 1) m = fmaxf(m, __shfl_xor(m, off));
        float e0 = (lane < 49)      ? __expf(v0 - m) : 0.0f;
        float e1 = (lane + 32 < 49) ? __expf(v1 - m) : 0.0f;
        float s = e0 + e1;
        #pragma unroll
        for (int off = 16; off > 0; off >>= 1) s += __shfl_xor(s, off);
        float inv = 1.0f / s;
        if (lane < 49)      s_alpha[lane]      = e0 * inv;
        if (lane + 32 < 49) s_alpha[lane + 32] = e1 * inv;
    }
    __syncthreads();

    const float* fb = features + (long)b * 49 * 2048;
    for (int d = tid; d < 2048; d += 256) {
        float acc = 0.0f;
        #pragma unroll 7
        for (int n = 0; n < 49; ++n) acc += s_alpha[n] * fb[n * 2048 + d];
        xcat[(long)b * 3072 + 512 + d] = (bf16)acc;
    }
}

__global__ void k_assemble_x(const bf16* __restrict__ embT, const float* __restrict__ h,
                             bf16* __restrict__ xcat, int t)
{
    int i = blockIdx.x * blockDim.x + threadIdx.x;  // 128*1024
    if (i >= 128 * 1024) return;
    int b = i >> 10, j = i & 1023;
    if (j < 512) {
        xcat[(long)b * 3072 + j] = embT[((long)t * 128 + b) * 512 + j];
    } else {
        int k = j - 512;
        xcat[(long)b * 3072 + 2560 + k] = (bf16)h[b * 512 + k];
    }
}

__device__ __forceinline__ float sigmoidf_(float x) { return 1.0f / (1.0f + __expf(-x)); }

__global__ void k_lstm(const float* __restrict__ gates, float* __restrict__ h,
                       float* __restrict__ c, bf16* __restrict__ hbf)
{
    int i = blockIdx.x * blockDim.x + threadIdx.x;  // 128*512
    if (i >= 128 * 512) return;
    int b = i >> 9, j = i & 511;
    const float* g = gates + (long)b * 2048;
    float ig = sigmoidf_(g[j]);
    float fg = sigmoidf_(g[512 + j]);
    float gg = tanhf(g[1024 + j]);
    float og = sigmoidf_(g[1536 + j]);
    float cn = fg * c[i] + ig * gg;
    float hn = og * tanhf(cn);
    c[i] = cn; h[i] = hn; hbf[i] = (bf16)hn;
}

__global__ void k_copy_hc(const float* __restrict__ h, const float* __restrict__ c,
                          float* __restrict__ out)
{
    int i = blockIdx.x * blockDim.x + threadIdx.x;
    if (i < 128 * 512) { out[i] = h[i]; out[65536 + i] = c[i]; }
}

// ---------------------------------------------------------------------------
extern "C" void kernel_launch(void* const* d_in, const int* in_sizes, int n_in,
                              void* d_out, int out_size, void* d_ws, size_t ws_size,
                              hipStream_t stream)
{
    (void)in_sizes; (void)n_in; (void)out_size; (void)ws_size;

    const float* features    = (const float*)d_in[0];
    const int*   captions    = (const int*)  d_in[1];
    const float* embed_w     = (const float*)d_in[2];
    const float* W_attn_feat = (const float*)d_in[3];
    const float* W_attn_h    = (const float*)d_in[4];
    const float* w_score     = (const float*)d_in[5];
    const float* W_ih        = (const float*)d_in[6];
    const float* W_hh        = (const float*)d_in[7];
    const float* b_ih        = (const float*)d_in[8];
    const float* b_hh        = (const float*)d_in[9];
    const float* W_out       = (const float*)d_in[10];
    const float* b_out       = (const float*)d_in[11];

    const int B = 128, N = 49, D = 2048, H = 512, V = 10000, T = 32;
    const int KX = H + D + H;       // 3072 = [emb | ctx | h]
    const int Vpad = 80 * BN;       // 10240: pad W_out rows so B over-reads stay in-bounds

    // workspace carve-out (256B aligned)
    char* wsb = (char*)d_ws;
    size_t off = 0;
    auto alloc = [&](size_t bytes) -> char* {
        char* p = wsb + off;
        off = (off + bytes + 255) & ~(size_t)255;
        return p;
    };
    bf16*  featbf = (bf16*) alloc((size_t)B * N * D * 2);        // A for proj GEMM
    bf16*  wafeat = (bf16*) alloc((size_t)(H + BN) * D * 2);     // [N=H, K=D] (+row pad)
    bf16*  wattnh = (bf16*) alloc((size_t)(H + BN) * H * 2);     // [N=H, K=H] (+row pad)
    bf16*  wcat   = (bf16*) alloc((size_t)(4 * H + BN) * KX * 2);// [N=4H, K=3072]
    bf16*  woutb  = (bf16*) alloc((size_t)Vpad * H * 2);         // [N=V, K=H] (padded)
    bf16*  embT   = (bf16*) alloc((size_t)T * B * H * 2);
    float* biasg  = (float*)alloc((size_t)4 * H * 4);            // b_ih + b_hh
    float* proj   = (float*)alloc((size_t)B * N * H * 4);
    float* hbuf   = (float*)alloc((size_t)B * H * 4);
    float* cbuf   = (float*)alloc((size_t)B * H * 4);
    bf16*  hbf    = (bf16*) alloc((size_t)B * H * 2);
    float* phb    = (float*)alloc((size_t)B * H * 4);
    bf16*  xcat   = (bf16*) alloc((size_t)B * KX * 2);
    float* gates  = (float*)alloc((size_t)B * (4 * H) * 4);

    auto cdiv = [](long a, long b) { return (int)((a + b - 1) / b); };

    // conversions / gathers (deterministic, redone every call)
    { long tot = (long)B * N * D;     k_cvt_rows<<<cdiv(tot,256),256,0,stream>>>(features,    featbf, tot, D,     D,  0); }
    { long tot = (long)H * D;         k_cvt_rows<<<cdiv(tot,256),256,0,stream>>>(W_attn_feat, wafeat, tot, D,     D,  0); }
    { long tot = (long)H * H;         k_cvt_rows<<<cdiv(tot,256),256,0,stream>>>(W_attn_h,    wattnh, tot, H,     H,  0); }
    { long tot = (long)4*H * (H + D); k_cvt_rows<<<cdiv(tot,256),256,0,stream>>>(W_ih,        wcat,   tot, H + D, KX, 0); }
    { long tot = (long)4*H * H;       k_cvt_rows<<<cdiv(tot,256),256,0,stream>>>(W_hh,        wcat,   tot, H,     KX, H + D); }
    { long tot = (long)V * H;         k_cvt_rows<<<cdiv(tot,256),256,0,stream>>>(W_out,       woutb,  tot, H,     H,  0); }
    k_embed_gather<<<cdiv((long)T * B * H, 256), 256, 0, stream>>>(embed_w, captions, embT);
    k_bias_add<<<cdiv(4 * H, 256), 256, 0, stream>>>(b_ih, b_hh, biasg, 4 * H);
    k_zero_state<<<cdiv(B * H, 256), 256, 0, stream>>>(hbuf, cbuf, hbf, B * H);

    auto gemm = [&](const bf16* Am, int lda, const bf16* Bm, int ldb,
                    float* Cm, int ldc, const float* bs, int M_, int N_, int K_) {
        dim3 g(cdiv(N_, BN), cdiv(M_, BM));
        gemm_bf16_wmma<<<g, 256, 0, stream>>>(Am, lda, Bm, ldb, Cm, ldc, bs, M_, N_, K_);
    };

    // step-invariant: proj_feat = features @ W_attn_feat^T   [6272 x 512]
    gemm(featbf, D, wafeat, D, proj, H, nullptr, B * N, H, D);

    float* out_logits = (float*)d_out;
    for (int t = 0; t < T; ++t) {
        // ph = h @ W_attn_h^T
        gemm(hbf, H, wattnh, H, phb, H, nullptr, B, H, H);
        // attention: scores/softmax/ctx -> x_cat[:, 512:2560]
        k_attn<<<B, 256, 0, stream>>>(proj, phb, w_score, features, xcat);
        // x_cat[:, 0:512] = emb_t ; x_cat[:, 2560:3072] = h
        k_assemble_x<<<cdiv(B * 1024, 256), 256, 0, stream>>>(embT, hbuf, xcat, t);
        // gates = x_cat @ [W_ih|W_hh]^T + (b_ih + b_hh)
        gemm(xcat, KX, wcat, KX, gates, 4 * H, biasg, B, 4 * H, KX);
        // LSTM cell pointwise; refresh h/c and bf16 h
        k_lstm<<<cdiv(B * H, 256), 256, 0, stream>>>(gates, hbuf, cbuf, hbf);
        // logits_t = h_new @ W_out^T + b_out   -> d_out[:, t, :]
        gemm(hbf, H, woutb, H, out_logits + (long)t * V, T * V, b_out, B, V, H);
    }

    // outputs: logits [B,T,V] then h [B,H] then c [B,H]
    k_copy_hc<<<cdiv(B * H, 256), 256, 0, stream>>>(hbuf, cbuf, out_logits + (long)B * T * V);
}
